// Attention_18708877541655
// MI455X (gfx1250) — matrix operand order repaired and checked by reading, not compiled, not run on hardware
//
#include <hip/hip_runtime.h>
#include <hip/hip_bf16.h>
#include <stdint.h>

typedef __attribute__((ext_vector_type(16))) __bf16        v16bf;
typedef __attribute__((ext_vector_type(8)))  float         v8f;
typedef __attribute__((ext_vector_type(8)))  unsigned int  v8u;

#define B_DIM 32
#define S_DIM 2048
#define H_DIM 1024
#define LDS_PITCH 1032   // bf16 elems per staged row: 1024 + 8 pad (avoids 16-way bank conflict)

__device__ __forceinline__ unsigned int pack_bf16(float lo, float hi) {
  unsigned int a = __float_as_uint(lo), b = __float_as_uint(hi);
  a = (a + 0x7FFFu + ((a >> 16) & 1u)) >> 16;          // round-to-nearest-even
  b = (b + 0x7FFFu + ((b >> 16) & 1u)) >> 16;
  return a | (b << 16);
}

// CDNA5 has a hardware transcendental V_TANH_F32 (listed as a TRANS32 op in the
// ISA's S_DELAY_ALU section). Use it when the builtin exists; ocml tanhf's
// polynomial expansion otherwise bloats the epilogue by ~30 VALU ops per call.
__device__ __forceinline__ float fast_tanh(float x) {
#if __has_builtin(__builtin_amdgcn_tanhf)
  return __builtin_amdgcn_tanhf(x);
#else
  return tanhf(x);
#endif
}

// ---------------------------------------------------------------------------
// Kernel 0: convert W (fp32 [1024][2048]) into bf16, pre-swizzled into the
// WMMA B-fragment layout. Index = ((((half*64+ht)*32+kc)*32+lane)*8+j) dwords.
// B frag (32x16 bf16): lane<16 -> K=0..15 (vgpr j: K=2j,2j+1), lane>=16 -> K=16..31.
// half 0 = W[:, 0:1024] (hidden part), half 1 = W[:, 1024:2048] (encoder part).
// ---------------------------------------------------------------------------
__global__ void __launch_bounds__(256)
wswz_kernel(const float* __restrict__ W, unsigned int* __restrict__ Wswz) {
  int i    = blockIdx.x * blockDim.x + threadIdx.x;   // 1,048,576 dwords total
  int j    = i & 7;
  int lane = (i >> 3) & 31;
  int kc   = (i >> 8) & 31;
  int ht   = (i >> 13) & 63;
  int half = (i >> 19) & 1;
  int n  = lane & 15;
  int k0 = ((lane < 16) ? 0 : 16) + 2 * j;
  int h  = ht * 16 + n;
  int kg = kc * 32 + k0;
  const float* src = W + (size_t)h * (2 * H_DIM) + half * H_DIM + kg;
  Wswz[i] = pack_bf16(src[0], src[1]);
}

// ---------------------------------------------------------------------------
// Stage 16 fp32 rows (1024 wide) into LDS as bf16 with padded pitch.
// ---------------------------------------------------------------------------
__device__ __forceinline__ void stage_rows_bf16(const float* __restrict__ src_base,
                                                size_t row_stride,
                                                unsigned short* __restrict__ aTile,
                                                int tid) {
  int r = tid >> 4, seg = tid & 15;
  const float* src = src_base + (size_t)r * row_stride + seg * 64;
  unsigned int* dst = (unsigned int*)&aTile[r * LDS_PITCH + seg * 64];
#pragma unroll
  for (int i = 0; i < 16; ++i) {
    float4 f = ((const float4*)src)[i];
    dst[i * 2 + 0] = pack_bf16(f.x, f.y);
    dst[i * 2 + 1] = pack_bf16(f.z, f.w);
  }
}

// A frag (16x32 bf16, contiguous, matches B): lane<16 -> K base 0, lane>=16 -> K base 16;
// vgpr j: K = base+2j, base+2j+1.
__device__ __forceinline__ v16bf load_a_frag(const unsigned int* __restrict__ arow,
                                             int kc, int hi) {
  v8u au;
  int kb = kc * 32 + (hi ? 16 : 0);
#pragma unroll
  for (int j = 0; j < 8; ++j) {
    int k0 = 2 * j;
    au[j] = arow[(kb + k0) >> 1];
  }
  return __builtin_bit_cast(v16bf, au);
}

// ---------------------------------------------------------------------------
// Kernel 1: u[b,h] = hidden[b] . W[h, 0:1024] + bias[h]   (WMMA, tiny GEMM)
// grid (8 col-chunks of 128, 2 row-tiles of 16), 256 threads (8 waves).
// ---------------------------------------------------------------------------
__global__ void __launch_bounds__(256)
u_kernel(const float* __restrict__ hidden, const unsigned int* __restrict__ Wswz,
         const float* __restrict__ bias, float* __restrict__ u_out) {
  __shared__ __align__(16) unsigned short aTile[16 * LDS_PITCH];
  int tid = threadIdx.x;
  int rowBase = blockIdx.y * 16;
  stage_rows_bf16(hidden + (size_t)rowBase * H_DIM, H_DIM, aTile, tid);
  __syncthreads();

  int wave = tid >> 5, lane = tid & 31;
  int m = lane & 15, hi = (lane >= 16);
  int ht = blockIdx.x * 8 + wave;                 // this wave's 16 output cols
  const unsigned int* arow = (const unsigned int*)&aTile[m * LDS_PITCH];
  const unsigned int* bbase = Wswz + ((size_t)ht * 32 * 32 + lane) * 8;  // half 0

  v8f acc = {};
  for (int kc = 0; kc < 32; ++kc) {
    v16bf av = load_a_frag(arow, kc, hi);
    const unsigned int* bp = bbase + (size_t)kc * 256;
    uint4 q0 = *(const uint4*)bp;
    uint4 q1 = *(const uint4*)(bp + 4);
    v8u bu = {q0.x, q0.y, q0.z, q0.w, q1.x, q1.y, q1.z, q1.w};
    acc = __builtin_amdgcn_wmma_f32_16x16x32_bf16(
        false, av, false, __builtin_bit_cast(v16bf, bu), (short)0, acc, false, false);
  }
  int col = ht * 16 + m;
  float bv = bias[col];
#pragma unroll
  for (int r = 0; r < 8; ++r) {
    int row = rowBase + r + hi * 8;               // C/D layout: M = r (+8 for hi half)
    u_out[(size_t)row * H_DIM + col] = acc[r] + bv;
  }
}

// ---------------------------------------------------------------------------
// Kernel 2: main GEMM + tanh + v-dot + mask -> scores[b][s]
// grid (S/16, B) ; per WG: 16 s-rows of one batch b, full N=1024, K=1024.
// 8 waves x 8 N-tiles x 32 K-chunks of v_wmma_f32_16x16x32_bf16.
// ---------------------------------------------------------------------------
__global__ void __launch_bounds__(256)
attn_scores_kernel(const float* __restrict__ eo, const unsigned int* __restrict__ Wswz,
                   const float* __restrict__ u_in, const float* __restrict__ v_in,
                   const int* __restrict__ mask, float* __restrict__ scores) {
  __shared__ __align__(16) unsigned short aTile[16 * LDS_PITCH];
  __shared__ float red[8 * 16];
  int tid = threadIdx.x;
  int b = blockIdx.y;
  int sBase = blockIdx.x * 16;

  // stage eo[sBase..sBase+15, b, :] -> bf16 LDS (row stride between s is B*H)
  stage_rows_bf16(eo + ((size_t)sBase * B_DIM + b) * H_DIM,
                  (size_t)B_DIM * H_DIM, aTile, tid);
  __syncthreads();

  int wave = tid >> 5, lane = tid & 31;
  int m = lane & 15, hi = (lane >= 16);
  const unsigned int* arow = (const unsigned int*)&aTile[m * LDS_PITCH];

  v8f acc[8];
#pragma unroll
  for (int nt = 0; nt < 8; ++nt) acc[nt] = (v8f){};

  for (int kc = 0; kc < 32; ++kc) {
    v16bf av = load_a_frag(arow, kc, hi);
#pragma unroll
    for (int nt = 0; nt < 8; ++nt) {
      // half 1 of Wswz: global ht = 64 + wave*8 + nt
      const unsigned int* bp =
          Wswz + ((((size_t)(64 + wave * 8 + nt) * 32 + kc) * 32) + lane) * 8;
      uint4 q0 = *(const uint4*)bp;
      uint4 q1 = *(const uint4*)(bp + 4);
      v8u bu = {q0.x, q0.y, q0.z, q0.w, q1.x, q1.y, q1.z, q1.w};
      acc[nt] = __builtin_amdgcn_wmma_f32_16x16x32_bf16(
          false, av, false, __builtin_bit_cast(v16bf, bu), (short)0, acc[nt], false, false);
    }
  }

  // epilogue: score contribution = tanh(acc + u[b,col]) * v[col]
  float rowSum[8];
#pragma unroll
  for (int r = 0; r < 8; ++r) rowSum[r] = 0.0f;
  const float* ub = u_in + (size_t)b * H_DIM;
#pragma unroll
  for (int nt = 0; nt < 8; ++nt) {
    int col = wave * 128 + nt * 16 + m;
    float uv = ub[col];
    float vv = v_in[col];
#pragma unroll
    for (int r = 0; r < 8; ++r) rowSum[r] += fast_tanh(acc[nt][r] + uv) * vv;
  }
  // reduce over the 16 lanes of each half-wave (xor 1,2,4,8 stays in-half)
#pragma unroll
  for (int off = 1; off < 16; off <<= 1) {
#pragma unroll
    for (int r = 0; r < 8; ++r) rowSum[r] += __shfl_xor(rowSum[r], off, 32);
  }
  if (m == 0) {
#pragma unroll
    for (int r = 0; r < 8; ++r) red[wave * 16 + hi * 8 + r] = rowSum[r];
  }
  __syncthreads();
  if (tid < 16) {
    float tot = 0.0f;
#pragma unroll
    for (int w = 0; w < 8; ++w) tot += red[w * 16 + tid];
    int s = sBase + tid;
    float val = (mask[(size_t)b * S_DIM + s] != 0) ? -1e9f : tot;  // where(mask, -1e9, x)
    scores[(size_t)b * S_DIM + s] = val;
  }
}

// ---------------------------------------------------------------------------
// Kernel 3: masked softmax over S per batch row.
// ---------------------------------------------------------------------------
__global__ void __launch_bounds__(256)
softmax_kernel(const float* __restrict__ scores, float* __restrict__ out) {
  __shared__ float red[256];
  int b = blockIdx.x, tid = threadIdx.x;
  const float* row = scores + (size_t)b * S_DIM;
  float vals[8];
  float mx = -3.4e38f;
#pragma unroll
  for (int i = 0; i < 8; ++i) { vals[i] = row[tid + i * 256]; mx = fmaxf(mx, vals[i]); }
  red[tid] = mx; __syncthreads();
  for (int off = 128; off > 0; off >>= 1) {
    if (tid < off) red[tid] = fmaxf(red[tid], red[tid + off]);
    __syncthreads();
  }
  mx = red[0]; __syncthreads();
  float sum = 0.0f;
#pragma unroll
  for (int i = 0; i < 8; ++i) { vals[i] = __expf(vals[i] - mx); sum += vals[i]; }
  red[tid] = sum; __syncthreads();
  for (int off = 128; off > 0; off >>= 1) {
    if (tid < off) red[tid] += red[tid + off];
    __syncthreads();
  }
  float inv = 1.0f / red[0];
#pragma unroll
  for (int i = 0; i < 8; ++i) out[(size_t)b * S_DIM + tid + i * 256] = vals[i] * inv;
}

// ---------------------------------------------------------------------------
extern "C" void kernel_launch(void* const* d_in, const int* in_sizes, int n_in,
                              void* d_out, int out_size, void* d_ws, size_t ws_size,
                              hipStream_t stream) {
  const float* hidden = (const float*)d_in[0];        // [B,H]
  const float* eo     = (const float*)d_in[1];        // [S,B,H]
  const int*   mask   = (const int*)d_in[2];          // [B,1,S] bool->int
  const float* W      = (const float*)d_in[3];        // [H,2H]
  const float* bias   = (const float*)d_in[4];        // [H]
  const float* v      = (const float*)d_in[5];        // [H]
  float* out = (float*)d_out;                         // [B,1,S] fp32

  unsigned char* ws = (unsigned char*)d_ws;
  unsigned int* Wswz  = (unsigned int*)ws;                               // 4 MB
  float* u_ws   = (float*)(ws + (size_t)4 * 1024 * 1024);                // 128 KB
  float* scores = (float*)(ws + (size_t)4 * 1024 * 1024 + 128 * 1024);   // 256 KB

  wswz_kernel<<<4096, 256, 0, stream>>>(W, Wswz);
  u_kernel<<<dim3(8, 2), 256, 0, stream>>>(hidden, Wswz, bias, u_ws);
  attn_scores_kernel<<<dim3(S_DIM / 16, B_DIM), 256, 0, stream>>>(eo, Wswz, u_ws, v, mask, scores);
  softmax_kernel<<<B_DIM, 256, 0, stream>>>(scores, out);
}
